// GraphVAE_58866821759499
// MI455X (gfx1250) — matrix operand order repaired and checked by reading, not compiled, not run on hardware
//
#include <hip/hip_runtime.h>
#include <stdint.h>
#include <stddef.h>

// ---------------------------------------------------------------------------
// Types for CDNA5 WMMA (wave32, 16x16x32 bf16 -> f32)
// ---------------------------------------------------------------------------
typedef __bf16 bf16_t;
typedef bf16_t v16bf __attribute__((ext_vector_type(16)));
typedef float  v8f   __attribute__((ext_vector_type(8)));

#define NODE_C 64   // hidden / latent width (H == L == 64)
#define GRAPHS 128

// ---------------------------------------------------------------------------
// Small utility kernels
// ---------------------------------------------------------------------------
__global__ void k_zero_f32(float* __restrict__ p, long long n) {
    long long i = (long long)blockIdx.x * blockDim.x + threadIdx.x;
    if (i < n) p[i] = 0.0f;
}

// deg[d] += 1 over edge destinations
__global__ void k_degree(const int* __restrict__ dst, float* __restrict__ deg, long long E) {
    long long e = (long long)blockIdx.x * blockDim.x + threadIdx.x;
    if (e < E) atomicAdd(&deg[dst[e]], 1.0f);
}

// deg -> rsqrt(deg + 1)   (self-loop guarantees > 0)
__global__ void k_dinv(float* __restrict__ deg, int N) {
    int n = blockIdx.x * blockDim.x + threadIdx.x;
    if (n < N) deg[n] = rsqrtf(deg[n] + 1.0f);
}

// f32 -> bf16 weight pack (row-major preserved)
__global__ void k_pack_bf16(const float* __restrict__ W, bf16_t* __restrict__ Wb, int n) {
    int i = blockIdx.x * blockDim.x + threadIdx.x;
    if (i < n) Wb[i] = (bf16_t)W[i];
}

// ---------------------------------------------------------------------------
// GEMM: C[M,64] = A[M,K] @ Wb[K,64]   (A f32 -> bf16 on the fly, Wb bf16)
//  * Weights staged once per block in LDS (K*64 bf16 = 32KB @ K=256).
//  * One wave -> one 16-row M tile, all 4 N tiles (A fragment reused x4).
//  * K templated -> k-loop fully unrolled, WMMA stream schedulable.
//  * 256 threads/block = 8 waves = 128 rows per block.
// ---------------------------------------------------------------------------
template <int K>
__global__ __launch_bounds__(256)
void k_gemm_bf16(const float* __restrict__ A, const bf16_t* __restrict__ Wb,
                 float* __restrict__ C, int M) {
    __shared__ bf16_t smem[K * NODE_C];

    const unsigned tid  = threadIdx.x;
    const unsigned lane = tid & 31u;
    const unsigned wave = tid >> 5;

    // ---- cooperative weight stage: global -> LDS (16B vectors) ----
    {
        const uint4* gsrc = (const uint4*)Wb;
        uint4*       ldst = (uint4*)smem;
        const int nvec = (K * NODE_C * 2) / 16;
#pragma unroll 4
        for (int i = (int)tid; i < nvec; i += 256) ldst[i] = gsrc[i];
    }
    __syncthreads();

    const int mtile = (int)(blockIdx.x * 8u + wave);
    if (mtile * 16 >= M) return;                   // wave-uniform: EXEC stays all-1s

    const int row = mtile * 16 + (int)(lane & 15u);   // A row for this lane
    const int kb  = (int)(lane >> 4u) * 8;            // K sub-chunk base (0 or 8)

    v8f c0 = {}; v8f c1 = {}; v8f c2 = {}; v8f c3 = {};

#pragma unroll
    for (int ks = 0; ks < (K >> 5); ++ks) {
        // ---- A fragment: row `row`, K = ks*32 + {kb..kb+7, 16+kb..16+kb+7}
        const float* ap = A + (size_t)row * K + ks * 32 + kb;
        v16bf a;
#pragma unroll
        for (int e = 0; e < 8; ++e) {
            a[e]     = (bf16_t)ap[e];
            a[e + 8] = (bf16_t)ap[e + 16];
        }
        // ---- B fragments from LDS: col = lane&15 per n-tile, K = ks*32 + {kb..kb+7, 16+kb..16+kb+7}
        const bf16_t* brow = smem + (size_t)(ks * 32 + kb) * NODE_C + (int)(lane & 15u);
        v16bf b0, b1, b2, b3;
#pragma unroll
        for (int e = 0; e < 8; ++e) {
            const bf16_t* p0 = brow + (size_t)e * NODE_C;
            const bf16_t* p1 = brow + (size_t)(e + 16) * NODE_C;
            b0[e] = p0[0];  b0[e + 8] = p1[0];
            b1[e] = p0[16]; b1[e + 8] = p1[16];
            b2[e] = p0[32]; b2[e + 8] = p1[32];
            b3[e] = p0[48]; b3[e + 8] = p1[48];
        }

        c0 = __builtin_amdgcn_wmma_f32_16x16x32_bf16(false, a, false, b0, (short)0, c0, false, false);
        c1 = __builtin_amdgcn_wmma_f32_16x16x32_bf16(false, a, false, b1, (short)0, c1, false, false);
        c2 = __builtin_amdgcn_wmma_f32_16x16x32_bf16(false, a, false, b2, (short)0, c2, false, false);
        c3 = __builtin_amdgcn_wmma_f32_16x16x32_bf16(false, a, false, b3, (short)0, c3, false, false);
    }

    // ---- store: VGPR j -> rows (mtile*16 + j + 8*(lane>=16)), col = lane&15
    const int rbase = mtile * 16 + (int)((lane >> 4u) << 3u);
    const int col   = (int)(lane & 15u);
#pragma unroll
    for (int j = 0; j < 8; ++j) {
        float* orow = C + (size_t)(rbase + j) * NODE_C + col;
        orow[0]  = c0[j];
        orow[16] = c1[j];
        orow[32] = c2[j];
        orow[48] = c3[j];
    }
}

// ---------------------------------------------------------------------------
// GCN conv pieces: acc = dinv^2 * h  (self loop), then edge scatter-add,
// then bias (+ optional relu) in place.
// ---------------------------------------------------------------------------
__global__ void k_self_init(const float* __restrict__ h, const float* __restrict__ dinv,
                            float* __restrict__ acc, long long NC) {
    long long i = (long long)blockIdx.x * blockDim.x + threadIdx.x;
    if (i < NC) {
        int n = (int)(i >> 6);
        float dv = dinv[n];
        acc[i] = dv * dv * h[i];
    }
}

// 16 lanes per edge, float4 per lane: acc[dst, c..c+3] += dinv[s]*dinv[d]*h[src, c..c+3]
__global__ void k_edge_scatter(const int* __restrict__ src, const int* __restrict__ dst,
                               const float* __restrict__ dinv, const float* __restrict__ h,
                               float* __restrict__ acc, long long E) {
    long long t = (long long)blockIdx.x * blockDim.x + threadIdx.x;
    long long e = t >> 4;
    int c = (int)(t & 15) * 4;
    if (e < E) {
        int s = src[e], d = dst[e];
        float nrm = dinv[s] * dinv[d];
        const float4 hv = *(const float4*)(h + (((size_t)s) << 6) + c);
        float* ap = acc + (((size_t)d) << 6) + c;
        atomicAdd(ap + 0, nrm * hv.x);
        atomicAdd(ap + 1, nrm * hv.y);
        atomicAdd(ap + 2, nrm * hv.z);
        atomicAdd(ap + 3, nrm * hv.w);
        // warm a future edge's source row (emits global_prefetch_b8)
        long long ef = e + 2048;
        if (ef < E) {
            int sf = src[ef];
            __builtin_prefetch(h + (((size_t)sf) << 6) + c, 0, 1);
        }
    }
}

__global__ void k_finish(float* __restrict__ acc, const float* __restrict__ bias,
                         long long NC, int do_relu) {
    long long i = (long long)blockIdx.x * blockDim.x + threadIdx.x;
    if (i < NC) {
        float v = acc[i] + bias[(int)(i & 63)];
        acc[i] = do_relu ? fmaxf(v, 0.0f) : v;
    }
}

// ---------------------------------------------------------------------------
// Mean pool over sorted batch assignment
// ---------------------------------------------------------------------------
__global__ void k_pool_scatter(const int* __restrict__ batch, const float* __restrict__ z,
                               float* __restrict__ zsum, float* __restrict__ cnt, long long NC) {
    long long i = (long long)blockIdx.x * blockDim.x + threadIdx.x;
    if (i < NC) {
        int n = (int)(i >> 6);
        int c = (int)(i & 63);
        int g = batch[n];
        atomicAdd(&zsum[(size_t)g * NODE_C + c], z[i]);
        if (c == 0) atomicAdd(&cnt[g], 1.0f);
    }
}

__global__ void k_pool_div(const float* __restrict__ zsum, const float* __restrict__ cnt,
                           float* __restrict__ zp, int GC) {
    int i = blockIdx.x * blockDim.x + threadIdx.x;
    if (i < GC) {
        int g = i >> 6;
        zp[i] = zsum[i] / fmaxf(cnt[g], 1.0f);
    }
}

// ---------------------------------------------------------------------------
// Decoder (tiny): hid = relu(zp @ Wd1 + bd1); x_hat = hid @ Wd2 + bd2
// ---------------------------------------------------------------------------
__global__ void k_dec1(const float* __restrict__ zp, const float* __restrict__ Wd1,
                       const float* __restrict__ bd1, float* __restrict__ hid) {
    int g = blockIdx.x, j = threadIdx.x;   // 128 x 64
    float s = bd1[j];
#pragma unroll 8
    for (int l = 0; l < 64; ++l) s += zp[g * 64 + l] * Wd1[l * 64 + j];
    hid[g * 64 + j] = fmaxf(s, 0.0f);
}

__global__ void k_dec2(const float* __restrict__ hid, const float* __restrict__ Wd2,
                       const float* __restrict__ bd2, float* __restrict__ xhat) {
    int g = blockIdx.x, d = threadIdx.x;   // 128 x 256
    float s = bd2[d];
#pragma unroll 8
    for (int j = 0; j < 64; ++j) s += hid[g * 64 + j] * Wd2[j * 256 + d];
    xhat[g * 256 + d] = s;
}

// ---------------------------------------------------------------------------
// Host launcher
// ---------------------------------------------------------------------------
extern "C" void kernel_launch(void* const* d_in, const int* in_sizes, int n_in,
                              void* d_out, int out_size, void* d_ws, size_t ws_size,
                              hipStream_t stream) {
    (void)n_in; (void)out_size; (void)ws_size;

    const float* x      = (const float*)d_in[0];
    const int*   eidx   = (const int*)d_in[1];
    const int*   batch  = (const int*)d_in[2];
    const float* W1     = (const float*)d_in[3];
    const float* b1     = (const float*)d_in[4];
    const float* W2     = (const float*)d_in[5];
    const float* b2     = (const float*)d_in[6];
    const float* Wd1    = (const float*)d_in[7];
    const float* bd1    = (const float*)d_in[8];
    const float* Wd2    = (const float*)d_in[9];
    const float* bd2    = (const float*)d_in[10];

    const int D = 256;
    const int N = in_sizes[0] / D;            // 100000
    const long long E = in_sizes[1] / 2;      // 3200000
    const int* esrc = eidx;
    const int* edst = eidx + E;

    // ---- workspace layout (256B aligned slices) ----
    char* wsp = (char*)d_ws;
    size_t off = 0;
    auto alloc = [&](size_t bytes) -> char* {
        char* p = wsp + off;
        off += (bytes + 255) & ~(size_t)255;
        return p;
    };
    float*  dinv = (float*)alloc((size_t)N * 4);
    bf16_t* Wb1  = (bf16_t*)alloc((size_t)D * NODE_C * 2);
    bf16_t* Wb2  = (bf16_t*)alloc((size_t)NODE_C * NODE_C * 2);
    float*  bufA = (float*)alloc((size_t)N * NODE_C * 4);  // lin1 out h, then lin2 out
    float*  bufB = (float*)alloc((size_t)N * NODE_C * 4);  // conv1 acc -> relu(h)
    float*  bufC = (float*)alloc((size_t)N * NODE_C * 4);  // conv2 acc -> z
    float*  zsum = (float*)alloc((size_t)GRAPHS * NODE_C * 4);
    float*  cnt  = (float*)alloc((size_t)GRAPHS * 4);
    float*  hid  = (float*)alloc((size_t)GRAPHS * NODE_C * 4);

    float* out_xhat = (float*)d_out;                       // [128, 256]
    float* out_zp   = out_xhat + (size_t)GRAPHS * D;       // [128, 64]

    const long long NC = (long long)N * NODE_C;
    const unsigned TB = 256;
    auto blocks = [&](long long work) -> unsigned { return (unsigned)((work + TB - 1) / TB); };

    // 1) symmetric degree normalization: dinv = rsqrt(deg_in + 1)
    k_zero_f32<<<blocks(N), TB, 0, stream>>>(dinv, N);
    k_degree<<<blocks(E), TB, 0, stream>>>(edst, dinv, E);
    k_dinv<<<blocks(N), TB, 0, stream>>>(dinv, N);

    // 2) pack weights to bf16 for WMMA
    k_pack_bf16<<<blocks(D * NODE_C), TB, 0, stream>>>(W1, Wb1, D * NODE_C);
    k_pack_bf16<<<blocks(NODE_C * NODE_C), TB, 0, stream>>>(W2, Wb2, NODE_C * NODE_C);

    const unsigned gemm_blocks = (unsigned)(((N + 15) / 16 + 7) / 8);

    // 3) conv1: h = x @ W1 ; acc = dinv^2*h + scatter ; relu(acc + b1)
    k_gemm_bf16<256><<<gemm_blocks, TB, 0, stream>>>(x, Wb1, bufA, N);
    k_self_init<<<blocks(NC), TB, 0, stream>>>(bufA, dinv, bufB, NC);
    k_edge_scatter<<<blocks(E * 16), TB, 0, stream>>>(esrc, edst, dinv, bufA, bufB, E);
    k_finish<<<blocks(NC), TB, 0, stream>>>(bufB, b1, NC, 1);

    // 4) conv2: hz = relu(h) @ W2 ; acc ; z = acc + b2
    k_gemm_bf16<64><<<gemm_blocks, TB, 0, stream>>>(bufB, Wb2, bufA, N);
    k_self_init<<<blocks(NC), TB, 0, stream>>>(bufA, dinv, bufC, NC);
    k_edge_scatter<<<blocks(E * 16), TB, 0, stream>>>(esrc, edst, dinv, bufA, bufC, E);
    k_finish<<<blocks(NC), TB, 0, stream>>>(bufC, b2, NC, 0);

    // 5) mean pool -> z_pool (written straight into d_out tail)
    k_zero_f32<<<blocks(GRAPHS * NODE_C), TB, 0, stream>>>(zsum, GRAPHS * NODE_C);
    k_zero_f32<<<1, GRAPHS, 0, stream>>>(cnt, GRAPHS);
    k_pool_scatter<<<blocks(NC), TB, 0, stream>>>(batch, bufC, zsum, cnt, NC);
    k_pool_div<<<blocks(GRAPHS * NODE_C), TB, 0, stream>>>(zsum, cnt, out_zp, GRAPHS * NODE_C);

    // 6) decoder -> x_hat
    k_dec1<<<GRAPHS, 64, 0, stream>>>(out_zp, Wd1, bd1, hid);
    k_dec2<<<GRAPHS, 256, 0, stream>>>(hid, Wd2, bd2, out_xhat);
}
